// CfaModel_76605036692282
// MI455X (gfx1250) — matrix operand order repaired and checked by reading, not compiled, not run on hardware
//
#include <hip/hip_runtime.h>

// ---------- problem constants ----------
#define C_DIM   1792
#define K_DIM   3136
#define ROWS    50176          // B*HW = 16*3136
#define M_TILE  128
#define N_TILE  64
#define CCHUNK  128
#define N_TILES 49             // 3136/64
#define NBLK    392            // 50176/128

#if defined(__has_builtin)
#if __has_builtin(__builtin_amdgcn_global_load_async_to_lds_b128)
#define HAVE_ASYNC_LDS 1
#endif
#endif

typedef __attribute__((ext_vector_type(16))) __bf16 v16bf;
typedef __attribute__((ext_vector_type(8)))  float  v8f;
typedef __attribute__((ext_vector_type(4)))  int    v4i;

#define GLOBAL_AS __attribute__((address_space(1)))
#define LDS_AS    __attribute__((address_space(3)))

union FragU { v16bf v; uint4 q[2]; };

__device__ __forceinline__ unsigned short f32_to_bf16(float f) {
  union { float f; unsigned u; } cv; cv.f = f;
  unsigned u = cv.u;
  u += 0x7FFFu + ((u >> 16) & 1u);       // round-to-nearest-even
  return (unsigned short)(u >> 16);
}

// pack two f32 -> two bf16 (round-half-up) in one v_perm_b32
__device__ __forceinline__ unsigned pack_bf16x2(float lo, float hi) {
  unsigned a = __float_as_uint(lo) + 0x8000u;
  unsigned b = __float_as_uint(hi) + 0x8000u;
  return __builtin_amdgcn_perm(b, a, 0x07060302);  // {b[31:16], a[31:16]}
}

// XOR-swizzled element offset inside a [rows][128] ushort LDS tile:
// 16B chunks permuted by row so ds_load_b128 fragment reads stay conflict-free.
__device__ __forceinline__ int sw(int row, int col) {
  int ch = (col >> 3) ^ (row & 15);
  return row * 128 + (ch << 3) + (col & 7);
}

__device__ __forceinline__ v8f wmma_bf16(v16bf a, v16bf b, v8f c) {
  return __builtin_amdgcn_wmma_f32_16x16x32_bf16(false, a, false, b, (short)0, c,
                                                 false, false);
}

#if HAVE_ASYNC_LDS
// CDNA5 async copy: 16B global -> LDS without VGPR round-trip (ASYNCcnt).
// Builtin parameter types (from hipcc diagnostic): v4i in AS1 / AS3.
__device__ __forceinline__ void async_copy_b128(const unsigned short* gp,
                                                unsigned short* lp) {
  __builtin_amdgcn_global_load_async_to_lds_b128(
      (GLOBAL_AS v4i*)(v4i*)(unsigned short*)gp,
      (LDS_AS v4i*)(v4i*)lp, 0, 0);
}
#endif

// sorted-ascending running top-6 (smallest) insert: 1 min + one bubble pass
__device__ __forceinline__ void insert6(float* t, float v) {
  t[5] = fminf(t[5], v);
#pragma unroll
  for (int i = 4; i >= 0; --i) {
    float lo = fminf(t[i], t[i + 1]);
    float hi = fmaxf(t[i], t[i + 1]);
    t[i] = lo; t[i + 1] = hi;
  }
}

// ---------- prepass 1: bank^T -> bf16 [K][C], plus c_sq ----------
__global__ __launch_bounds__(256) void cfa_bank_prep(
    const float* __restrict__ bank, unsigned short* __restrict__ bankT,
    float* __restrict__ csq) {
  const int k = blockIdx.x, t = threadIdx.x;
  float s = 0.f;
  for (int c = t; c < C_DIM; c += 256) {
    float x = bank[(size_t)c * K_DIM + k];
    s += x * x;
    bankT[(size_t)k * C_DIM + c] = f32_to_bf16(x);
  }
  __shared__ float red[256];
  red[t] = s; __syncthreads();
  for (int off = 128; off > 0; off >>= 1) {
    if (t < off) red[t] += red[t + off];
    __syncthreads();
  }
  if (t == 0) csq[k] = red[0];
}

// ---------- prepass 2: per-row ||f||^2 (4 rows / block, 64 threads / row) ----------
__global__ __launch_bounds__(256) void cfa_fsq_prep(
    const float* __restrict__ feat, float* __restrict__ fsq) {
  const int t = threadIdx.x, lt = t & 63, rl = t >> 6;
  const int row = blockIdx.x * 4 + rl;
  const float* p = feat + (size_t)row * C_DIM;
  float s = 0.f;
  for (int c = lt * 4; c < C_DIM; c += 256) {
    float4 v = *reinterpret_cast<const float4*>(p + c);
    s += v.x * v.x + v.y * v.y + v.z * v.z + v.w * v.w;
  }
  __shared__ float red[256];
  red[t] = s; __syncthreads();
  for (int off = 32; off > 0; off >>= 1) {
    if (lt < off) red[t] += red[t + off];
    __syncthreads();
  }
  if (lt == 0) fsq[row] = red[t];
}

// ---------- main fused GEMM + top-6 + hinge ----------
__global__ __launch_bounds__(256) void cfa_main(
    const float* __restrict__ feat, const unsigned short* __restrict__ bankT,
    const float* __restrict__ csq, const float* __restrict__ fsq,
    const float* __restrict__ radius, float* __restrict__ partials) {
  // staging tiles and the distance spill buffer have disjoint live ranges
  // (separated by the existing barriers) -> alias them. 48 KB total.
  // alignas(16) is required so uint4 LDS accesses lower to ds_load/store_b128.
  __shared__ alignas(16) union {
    struct {
      unsigned short A[M_TILE * CCHUNK];  // 32 KB
      unsigned short B[N_TILE * CCHUNK];  // 16 KB
    } st;
    float S[M_TILE * N_TILE];             // 32 KB
  } u;
  __shared__ float CsqS[N_TILE];

  const int tid  = threadIdx.x;
  const int lane = tid & 31, wid = tid >> 5;
  const int l15  = lane & 15, half = lane >> 4;
  const int m0w  = (wid & 3) * 32;                 // 4 waves along M
  const int n0w  = (wid >> 2) * 32;                // 2 waves along N
  const int rowBase = blockIdx.x * M_TILE;
  const float r2 = radius[0] * radius[0];

  float t6[6] = {3.0e38f, 3.0e38f, 3.0e38f, 3.0e38f, 3.0e38f, 3.0e38f};

  for (int nt = 0; nt < N_TILES; ++nt) {
    const int n0g = nt * N_TILE;
    if (tid < N_TILE) CsqS[tid] = csq[n0g + tid];

    const v8f zero = {0.f, 0.f, 0.f, 0.f, 0.f, 0.f, 0.f, 0.f};
    v8f acc[2][2] = {{zero, zero}, {zero, zero}};

    for (int c0 = 0; c0 < C_DIM; c0 += CCHUNK) {
      // stage A (f32 -> bf16 via v_perm), coalesced global, conflict-free LDS.
      // unroll 4: keep 4 loads in flight without exploding live VGPR ranges.
#pragma unroll 4
      for (int i = 0; i < 16; ++i) {
        int s  = tid + 256 * i;                   // 4096 float4 slots
        int r  = s >> 5;
        int c4 = (s & 31) << 2;
        float4 f4 = *reinterpret_cast<const float4*>(
            feat + (size_t)(rowBase + r) * C_DIM + c0 + c4);
        uint2 p;
        p.x = pack_bf16x2(f4.x, f4.y);
        p.y = pack_bf16x2(f4.z, f4.w);
        *reinterpret_cast<uint2*>(&u.st.A[sw(r, c4)]) = p;
      }
      // stage B (already bf16, pre-transposed [K][C])
#pragma unroll
      for (int i = 0; i < 4; ++i) {
        int s = tid + 256 * i;                    // 1024 uint4 slots
        int r = s >> 4;
        int q = (s & 15) << 3;
        const unsigned short* gp = bankT + (size_t)(n0g + r) * C_DIM + c0 + q;
        unsigned short* lp = &u.st.B[sw(r, q)];
#if HAVE_ASYNC_LDS
        async_copy_b128(gp, lp);
#else
        *reinterpret_cast<uint4*>(lp) = *reinterpret_cast<const uint4*>(gp);
#endif
      }
      if (c0 + CCHUNK < C_DIM)  // hint next A chunk toward L2 (global_prefetch_b8)
        __builtin_prefetch(feat + (size_t)(rowBase + (tid >> 1)) * C_DIM +
                               c0 + CCHUNK + (tid & 1) * 64, 0, 0);
#if HAVE_ASYNC_LDS
#if __has_builtin(__builtin_amdgcn_s_wait_asynccnt)
      __builtin_amdgcn_s_wait_asynccnt(0);
#else
      asm volatile("s_wait_asynccnt 0x0" ::: "memory");
#endif
#endif
      __syncthreads();

#pragma unroll
      for (int kk = 0; kk < CCHUNK; kk += 32) {
        FragU a0, a1, b0, b1;
        // A frag (16x32 bf16): lane-half holds K runs [0..7]/[16..23] or [8..15]/[24..31]
        {
          int ra = m0w + l15, rb = m0w + 16 + l15;
          a0.q[0] = *reinterpret_cast<const uint4*>(&u.st.A[sw(ra, kk + half * 8)]);
          a0.q[1] = *reinterpret_cast<const uint4*>(&u.st.A[sw(ra, kk + half * 8 + 16)]);
          a1.q[0] = *reinterpret_cast<const uint4*>(&u.st.A[sw(rb, kk + half * 8)]);
          a1.q[1] = *reinterpret_cast<const uint4*>(&u.st.A[sw(rb, kk + half * 8 + 16)]);
        }
        // B frag (32x16 bf16): lane-half holds K runs [0..7]/[16..23] or [8..15]/[24..31]
        {
          int ra = n0w + l15, rb = n0w + 16 + l15;
          b0.q[0] = *reinterpret_cast<const uint4*>(&u.st.B[sw(ra, kk + half * 8)]);
          b0.q[1] = *reinterpret_cast<const uint4*>(&u.st.B[sw(ra, kk + half * 8 + 16)]);
          b1.q[0] = *reinterpret_cast<const uint4*>(&u.st.B[sw(rb, kk + half * 8)]);
          b1.q[1] = *reinterpret_cast<const uint4*>(&u.st.B[sw(rb, kk + half * 8 + 16)]);
        }
        acc[0][0] = wmma_bf16(a0.v, b0.v, acc[0][0]);
        acc[0][1] = wmma_bf16(a0.v, b1.v, acc[0][1]);
        acc[1][0] = wmma_bf16(a1.v, b0.v, acc[1][0]);
        acc[1][1] = wmma_bf16(a1.v, b1.v, acc[1][1]);
      }
      __syncthreads();
    }

    // spill s = c_sq - 2*cross to LDS (D layout: M = v + 8*half, N = l15)
#pragma unroll
    for (int mi = 0; mi < 2; ++mi)
#pragma unroll
      for (int ni = 0; ni < 2; ++ni) {
        int col = n0w + 16 * ni + l15;
        float cs = CsqS[col];
#pragma unroll
        for (int v = 0; v < 8; ++v) {
          int r = m0w + 16 * mi + 8 * half + v;
          u.S[r * 64 + col] = cs - 2.0f * acc[mi][ni][v];
        }
      }
    __syncthreads();

    // running top-6 smallest; each thread owns half a row, skewed columns
    {
      int srow = tid & 127, sh = tid >> 7;
      for (int j = 0; j < 32; ++j) {
        int col = (sh << 5) + ((j + srow) & 31);
        insert6(t6, u.S[srow * 64 + col]);
      }
    }
    __syncthreads();   // protects aliased staging tiles of the next N-tile
  }

  // merge the two half-row top-6 lists, apply hinges
  const int srow = tid & 127, sh = tid >> 7;
  if (sh == 1) {
#pragma unroll
    for (int i = 0; i < 6; ++i) u.S[srow * 8 + i] = t6[i];
  }
  __syncthreads();
  float contrib = 0.f;
  if (sh == 0) {
#pragma unroll
    for (int i = 0; i < 6; ++i) insert6(t6, u.S[srow * 8 + i]);
    float fs = fsq[rowBase + srow];
#pragma unroll
    for (int i = 0; i < 3; ++i)            // attraction: vals[0:3] - r^2
      contrib += fmaxf(t6[i] + fs - r2, 0.f);
#pragma unroll
    for (int i = 3; i < 6; ++i)            // repulsion: r^2 - vals[3:6] - alpha
      contrib += fmaxf(r2 - (t6[i] + fs) - 0.1f, 0.f);
  }
  __syncthreads();
  u.S[tid] = contrib;                      // deterministic block reduction
  __syncthreads();
  for (int off = 128; off > 0; off >>= 1) {
    if (tid < off) u.S[tid] += u.S[tid + off];
    __syncthreads();
  }
  if (tid == 0) partials[blockIdx.x] = u.S[0];
}

// ---------- final reduction + scaling ----------
__global__ __launch_bounds__(256) void cfa_final(
    const float* __restrict__ partials, float* __restrict__ out) {
  __shared__ float red[256];
  int t = threadIdx.x;
  float s = 0.f;
  for (int i = t; i < NBLK; i += 256) s += partials[i];
  red[t] = s; __syncthreads();
  for (int off = 128; off > 0; off >>= 1) {
    if (t < off) red[t] += red[t + off];
    __syncthreads();
  }
  // (1/NU) * ( sum_att + sum_rep ) / (B*HW*3),  NU = 0.001
  if (t == 0) out[0] = red[0] * (1000.0f / 150528.0f);
}

extern "C" void kernel_launch(void* const* d_in, const int* in_sizes, int n_in,
                              void* d_out, int out_size, void* d_ws, size_t ws_size,
                              hipStream_t stream) {
  const float* feat   = (const float*)d_in[0];   // [16,3136,1792]
  const float* bank   = (const float*)d_in[1];   // [1792,3136]
  const float* radius = (const float*)d_in[2];   // [1]

  char* ws = (char*)d_ws;
  const size_t offBankT = 0;                                  // 11,239,424 B
  const size_t offCsq   = offBankT + (size_t)K_DIM * C_DIM * 2;
  const size_t offFsq   = offCsq + (size_t)K_DIM * 4;
  const size_t offPart  = offFsq + (size_t)ROWS * 4;

  unsigned short* bankT = (unsigned short*)(ws + offBankT);
  float* csq      = (float*)(ws + offCsq);
  float* fsq      = (float*)(ws + offFsq);
  float* partials = (float*)(ws + offPart);

  cfa_bank_prep<<<K_DIM, 256, 0, stream>>>(bank, bankT, csq);
  cfa_fsq_prep<<<ROWS / 4, 256, 0, stream>>>(feat, fsq);
  cfa_main<<<NBLK, 256, 0, stream>>>(feat, bankT, csq, fsq, radius, partials);
  cfa_final<<<1, 256, 0, stream>>>(partials, (float*)d_out);
}
